// WaveNetTemporalBlock_24644522345335
// MI455X (gfx1250) — matrix operand order repaired and checked by reading, not compiled, not run on hardware
//
#include <hip/hip_runtime.h>

// ---------------- problem constants (match reference) ----------------
#define NLAYER 9
#define BATCH  16
#define CCH    128
#define LSEQ   8192
#define KW     3
#define TL     64      // time-tile per workgroup

typedef __attribute__((ext_vector_type(16))) __bf16 bf16x16;
typedef __attribute__((ext_vector_type(8)))  float  f32x8;

union Frag16 { unsigned int u[8]; bf16x16 v; };

__device__ __forceinline__ unsigned short f2bf(float f) {
    union { float f; unsigned int u; } c; c.f = f;
    unsigned int u = c.u + 0x7FFFu + ((c.u >> 16) & 1u);   // round-to-nearest-even
    return (unsigned short)(u >> 16);
}

// ---------------- weight preprocessing: fp32 -> bf16 ----------------
// conv weights: (NL, O, I, K) fp32  ->  (NL, K, O, I) bf16  (i contiguous -> dword A loads)
__global__ void convert_conv_weights(const float* __restrict__ src,
                                     unsigned short* __restrict__ dst) {
    int idx = blockIdx.x * blockDim.x + threadIdx.x;
    const int total = NLAYER * KW * CCH * CCH;
    if (idx >= total) return;
    int i = idx % CCH;
    int o = (idx / CCH) % CCH;
    int k = (idx / (CCH * CCH)) % KW;
    int l =  idx / (CCH * CCH * KW);
    dst[idx] = f2bf(src[(((size_t)(l * CCH + o) * CCH) + i) * KW + k]);
}

// proj weights: (NL, O, C) fp32 -> same layout bf16 (c already contiguous)
__global__ void convert_proj_weights(const float* __restrict__ src,
                                     unsigned short* __restrict__ dst) {
    int idx = blockIdx.x * blockDim.x + threadIdx.x;
    const int total = NLAYER * CCH * CCH;
    if (idx >= total) return;
    dst[idx] = f2bf(src[idx]);
}

// ---------------- fused WaveNet layer ----------------
// one workgroup: batch b, 64 time positions, all 128 channels.
// LDS plan (128 KB total -> 2 workgroups per WGP on CDNA5's 320 KB):
//   xs   : 3 x 128 x 64 bf16 staged dilation taps            (48 KB)
//   wlds : 2 x 128 x 128 bf16 weights (f|g per tap, later s|r) (64 KB)
//   hs   : 128 x 64 bf16 gated activations                    (16 KB)
__global__ __launch_bounds__(256)
void wavenet_layer(const float* __restrict__ x_in,
                   float*       __restrict__ x_out,
                   float*       __restrict__ skip,
                   const unsigned short* __restrict__ wf_t,  // [3][128][128] bf16
                   const unsigned short* __restrict__ wg_t,  // [3][128][128] bf16
                   const unsigned short* __restrict__ wr_t,  // [128][128]    bf16
                   const unsigned short* __restrict__ ws_t,  // [128][128]    bf16
                   const float* __restrict__ bfv, const float* __restrict__ bgv,
                   const float* __restrict__ brv, const float* __restrict__ bsv,
                   int dil, int accumulate)
{
    __shared__ __align__(16) unsigned short xs[3 * CCH * TL];
    __shared__ __align__(16) unsigned short wlds[2 * CCH * CCH];
    __shared__ __align__(16) unsigned short hs[CCH * TL];

    const int b    = blockIdx.y;
    const int t0   = blockIdx.x * TL;
    const int tid  = threadIdx.x;
    const int lane = tid & 31;
    const int wave = tid >> 5;

    const size_t bbase = (size_t)b * CCH * LSEQ;
    const float* xb = x_in + bbase;

    // ---- stage 3 dilation taps of x into LDS as bf16 (coalesced in t) ----
    for (int row = tid; row < 3 * CCH; row += 256) {
        const int k = row / CCH;           // tap 0..2  (offset (k-1)*dil, pad = dil)
        const int c = row % CCH;
        const int tbase = t0 + (k - 1) * dil;
        const float* xr = xb + (size_t)c * LSEQ;
        unsigned short* dr = &xs[(size_t)row * TL];
        if (tbase >= 0 && tbase + TL <= LSEQ) {
            #pragma unroll 8
            for (int t = 0; t < TL; ++t) dr[t] = f2bf(xr[tbase + t]);
        } else {
            for (int t = 0; t < TL; ++t) {
                const int tg = tbase + t;
                dr[t] = (tg >= 0 && tg < LSEQ) ? f2bf(xr[tg]) : (unsigned short)0;
            }
        }
    }

    // ---- wave -> tile mapping: 8 waves cover 128(out) x 64(t) twice (f & g) ----
    const int nt  = wave & 3;        // N-tile: positions nt*16..+15
    const int mts = (wave >> 2) * 4; // 4 M-tiles: out-ch [mts*16, (mts+4)*16)
    const int hi  = lane >> 4;       // lane half
    const int mr  = lane & 15;       // A-matrix row within tile

    const f32x8 zero8 = {0.f,0.f,0.f,0.f,0.f,0.f,0.f,0.f};
    f32x8 facc[4], gacc[4];
    #pragma unroll
    for (int m = 0; m < 4; ++m) { facc[m] = zero8; gacc[m] = zero8; }

    // ---- dilated-conv GEMM: per tap, stage f+g weights to LDS, then 4 k-chunks ----
    for (int tap = 0; tap < KW; ++tap) {
        __syncthreads();   // xs staged (tap 0) / previous tap's LDS reads done
        {
            const uint4* s0 = (const uint4*)(wf_t + (size_t)tap * CCH * CCH);
            const uint4* s1 = (const uint4*)(wg_t + (size_t)tap * CCH * CCH);
            uint4* d0 = (uint4*)wlds;
            uint4* d1 = (uint4*)(wlds + CCH * CCH);
            #pragma unroll
            for (int it = 0; it < (CCH * CCH / 8) / 256; ++it) {
                const int i = it * 256 + tid;
                d0[i] = s0[i];
                d1[i] = s1[i];
            }
        }
        __syncthreads();

        #pragma unroll
        for (int kc = 0; kc < 4; ++kc) {
            // B fragment (32x16): lane%16 = N (t), elements = K in table order
            Frag16 bx;
            const unsigned short* xrow =
                &xs[((size_t)(tap * CCH + kc * 32)) * TL + nt * 16 + (lane & 15)];
            #pragma unroll
            for (int v = 0; v < 8; ++v) {
                const int kb = ((v >> 2) << 4) + (hi << 3) + ((v & 3) << 1);
                bx.u[v] = (unsigned int)xrow[(size_t)kb * TL] | ((unsigned int)xrow[(size_t)(kb + 1) * TL] << 16);
            }

            #pragma unroll
            for (int m = 0; m < 4; ++m) {
                const int o = (mts + m) * 16 + mr;
                const unsigned short* wfp = wlds + (size_t)o * CCH + kc * 32;
                const unsigned short* wgp = wfp + CCH * CCH;
                Frag16 af, ag;
                #pragma unroll
                for (int v = 0; v < 8; ++v) {
                    const int kk = ((v >> 2) << 4) + (hi << 3) + ((v & 3) << 1);
                    af.u[v] = *(const unsigned int*)(wfp + kk);
                    ag.u[v] = *(const unsigned int*)(wgp + kk);
                }
                facc[m] = __builtin_amdgcn_wmma_f32_16x16x32_bf16(
                              false, af.v, false, bx.v, (short)0, facc[m], false, false);
                gacc[m] = __builtin_amdgcn_wmma_f32_16x16x32_bf16(
                              false, ag.v, false, bx.v, (short)0, gacc[m], false, false);
            }
        }
    }

    // ---- gated activation, store h to LDS as bf16 ([c][t]) ----
    #pragma unroll
    for (int m = 0; m < 4; ++m) {
        #pragma unroll
        for (int r = 0; r < 8; ++r) {
            const int o = (mts + m) * 16 + r + 8 * hi;      // out channel
            const int tl = nt * 16 + (lane & 15);           // local t
            const float fv = facc[m][r] + bfv[o];
            const float gv = gacc[m][r] + bgv[o];
            const float h  = tanhf(fv) * (1.0f / (1.0f + __expf(-gv)));
            hs[(size_t)o * TL + tl] = f2bf(h);
        }
    }
    __syncthreads();   // conv LDS reads done + hs visible

    // ---- stage projection weights (skip | residual) into the same LDS region ----
    {
        const uint4* s0 = (const uint4*)ws_t;
        const uint4* s1 = (const uint4*)wr_t;
        uint4* d0 = (uint4*)wlds;
        uint4* d1 = (uint4*)(wlds + CCH * CCH);
        #pragma unroll
        for (int it = 0; it < (CCH * CCH / 8) / 256; ++it) {
            const int i = it * 256 + tid;
            d0[i] = s0[i];
            d1[i] = s1[i];
        }
    }
    __syncthreads();

    // ---- projection GEMMs (skip & residual), reduction = 4 chunks of 32 ----
    f32x8 sacc[4], racc[4];
    #pragma unroll
    for (int m = 0; m < 4; ++m) { sacc[m] = zero8; racc[m] = zero8; }

    #pragma unroll
    for (int kc = 0; kc < 4; ++kc) {
        Frag16 bh;
        const unsigned short* hrow = &hs[((size_t)(kc * 32)) * TL + nt * 16 + (lane & 15)];
        #pragma unroll
        for (int v = 0; v < 8; ++v) {
            const int kb = ((v >> 2) << 4) + (hi << 3) + ((v & 3) << 1);
            bh.u[v] = (unsigned int)hrow[(size_t)kb * TL] | ((unsigned int)hrow[(size_t)(kb + 1) * TL] << 16);
        }

        #pragma unroll
        for (int m = 0; m < 4; ++m) {
            const int o = (mts + m) * 16 + mr;
            const unsigned short* wsp = wlds + (size_t)o * CCH + kc * 32;
            const unsigned short* wrp = wsp + CCH * CCH;
            Frag16 as, ar;
            #pragma unroll
            for (int v = 0; v < 8; ++v) {
                const int kk = ((v >> 2) << 4) + (hi << 3) + ((v & 3) << 1);
                as.u[v] = *(const unsigned int*)(wsp + kk);
                ar.u[v] = *(const unsigned int*)(wrp + kk);
            }
            sacc[m] = __builtin_amdgcn_wmma_f32_16x16x32_bf16(
                          false, as.v, false, bh.v, (short)0, sacc[m], false, false);
            racc[m] = __builtin_amdgcn_wmma_f32_16x16x32_bf16(
                          false, ar.v, false, bh.v, (short)0, racc[m], false, false);
        }
    }

    // ---- epilogue: residual update + skip accumulation ----
    #pragma unroll
    for (int m = 0; m < 4; ++m) {
        #pragma unroll
        for (int r = 0; r < 8; ++r) {
            const int o = (mts + m) * 16 + r + 8 * hi;
            const int t = t0 + nt * 16 + (lane & 15);
            const size_t idx = bbase + (size_t)o * LSEQ + t;
            x_out[idx] = x_in[idx] + racc[m][r] + brv[o];
            const float sv = sacc[m][r] + bsv[o];
            skip[idx] = accumulate ? (skip[idx] + sv) : sv;
        }
    }
}

// ---------------- host-side orchestration ----------------
extern "C" void kernel_launch(void* const* d_in, const int* in_sizes, int n_in,
                              void* d_out, int out_size, void* d_ws, size_t ws_size,
                              hipStream_t stream) {
    const float* x  = (const float*)d_in[0];
    const float* wf = (const float*)d_in[1];
    const float* bf = (const float*)d_in[2];
    const float* wg = (const float*)d_in[3];
    const float* bg = (const float*)d_in[4];
    const float* wr = (const float*)d_in[5];
    const float* br = (const float*)d_in[6];
    const float* ws = (const float*)d_in[7];
    const float* bs = (const float*)d_in[8];
    float* skip = (float*)d_out;

    // scratch layout
    char* p = (char*)d_ws;
    const size_t convW = (size_t)NLAYER * KW * CCH * CCH;   // bf16 elems
    const size_t projW = (size_t)NLAYER * CCH * CCH;        // bf16 elems
    const size_t xElems = (size_t)BATCH * CCH * LSEQ;       // f32 elems
    unsigned short* wf_t = (unsigned short*)p; p += convW * 2;
    unsigned short* wg_t = (unsigned short*)p; p += convW * 2;
    unsigned short* wr_t = (unsigned short*)p; p += projW * 2;
    unsigned short* ws_t = (unsigned short*)p; p += projW * 2;
    float* xb0 = (float*)p; p += xElems * 4;
    float* xb1 = (float*)p;

    convert_conv_weights<<<(int)((convW + 255) / 256), 256, 0, stream>>>(wf, wf_t);
    convert_conv_weights<<<(int)((convW + 255) / 256), 256, 0, stream>>>(wg, wg_t);
    convert_proj_weights<<<(int)((projW + 255) / 256), 256, 0, stream>>>(wr, wr_t);
    convert_proj_weights<<<(int)((projW + 255) / 256), 256, 0, stream>>>(ws, ws_t);

    static const int dils[NLAYER] = {1, 2, 4, 8, 16, 32, 64, 128, 256};
    dim3 grid(LSEQ / TL, BATCH), block(256);
    float* bufs[2] = {xb0, xb1};
    const float* cur = x;
    for (int i = 0; i < NLAYER; ++i) {
        float* nxt = bufs[i & 1];
        const size_t cko = (size_t)i * KW * CCH * CCH;
        const size_t cpo = (size_t)i * CCH * CCH;
        wavenet_layer<<<grid, block, 0, stream>>>(
            cur, nxt, skip,
            wf_t + cko, wg_t + cko, wr_t + cpo, ws_t + cpo,
            bf + i * CCH, bg + i * CCH, br + i * CCH, bs + i * CCH,
            dils[i], i > 0 ? 1 : 0);
        cur = nxt;
    }
}
